// SparseBackpropController_75763223102098
// MI455X (gfx1250) — matrix operand order repaired and checked by reading, not compiled, not run on hardware
//
#include <hip/hip_runtime.h>
#include <cstdint>
#include <cstddef>

// ---------------------------------------------------------------------------
// Problem sizes (fixed by the reference)
// ---------------------------------------------------------------------------
#define B_SZ 4096
#define D_SZ 1024
#define H_SZ 2048
#define K_SZ 16

// GEMM tiling
#define BM 128
#define BN 128
#define BK 32
#define SA 40   // LDS row stride (halves), padded: 80B rows keep b128 chunks 16B-aligned
#define SB 40
#define NT (D_SZ / BK)

typedef __attribute__((ext_vector_type(16))) __bf16 v16bf;
typedef __attribute__((ext_vector_type(8)))  float  v8f;
typedef __attribute__((ext_vector_type(4)))  int    v4i;

#define AS1 __attribute__((address_space(1)))
#define AS3 __attribute__((address_space(3)))

__device__ __forceinline__ unsigned short f2bf(float f) {
  unsigned u = __float_as_uint(f);
  u = u + 0x7FFFu + ((u >> 16) & 1u);
  return (unsigned short)(u >> 16);
}
__device__ __forceinline__ float bf2f(unsigned short b) {
  return __uint_as_float(((unsigned)b) << 16);
}

__device__ __forceinline__ float fast_tanh(float x) {
#if __has_builtin(__builtin_amdgcn_tanhf)
  return __builtin_amdgcn_tanhf(x);
#elif __has_builtin(__builtin_amdgcn_tanh_f32)
  return __builtin_amdgcn_tanh_f32(x);
#else
  return tanhf(x);
#endif
}

// --- CDNA5 async global->LDS copy (ASYNCcnt path), with safe fallback -------
#if __has_builtin(__builtin_amdgcn_global_load_async_to_lds_b128)
#define HAVE_ASYNC_LDS 1
__device__ __forceinline__ void async_copy16(const unsigned short* g, unsigned short* l) {
  __builtin_amdgcn_global_load_async_to_lds_b128(
      (AS1 v4i*)(g), (AS3 v4i*)(l), 0, 0);
}
__device__ __forceinline__ void wait_async0() {
#if __has_builtin(__builtin_amdgcn_s_wait_asynccnt)
  __builtin_amdgcn_s_wait_asynccnt(0);
#else
  asm volatile("s_wait_asynccnt 0x0" ::: "memory");
#endif
}
#else
#define HAVE_ASYNC_LDS 0
__device__ __forceinline__ void async_copy16(const unsigned short* g, unsigned short* l) {
  *(uint4*)l = *(const uint4*)g;   // sync fallback: load + ds_store
}
__device__ __forceinline__ void wait_async0() {}
#endif

// ---------------------------------------------------------------------------
// Kernel 0: zero dense-W scratch and the per-row accumulator
// ---------------------------------------------------------------------------
__global__ void zero_ws(uint32_t* __restrict__ wd32, float* __restrict__ acc) {
  int i = blockIdx.x * blockDim.x + threadIdx.x;
  if (i < (H_SZ * D_SZ / 2)) wd32[i] = 0u;
  if (i < B_SZ) acc[i] = 0.0f;
}

// ---------------------------------------------------------------------------
// Kernel 1: convert x (f32) -> bf16 once; done 1x instead of per-N-block.
// ---------------------------------------------------------------------------
__global__ void convert_x(const float* __restrict__ x, unsigned short* __restrict__ xb) {
  int i = blockIdx.x * blockDim.x + threadIdx.x;   // one per 4 floats
  const float4 v = ((const float4*)x)[i];
  unsigned p0 = (unsigned)f2bf(v.x) | ((unsigned)f2bf(v.y) << 16);
  unsigned p1 = (unsigned)f2bf(v.z) | ((unsigned)f2bf(v.w) << 16);
  ((uint2*)xb)[i] = make_uint2(p0, p1);
}

// ---------------------------------------------------------------------------
// Kernel 2: scatter-accumulate w1 into dense bf16 W [H][D].
// One thread per row h -> sequential RMW handles duplicate indices, race-free.
// ---------------------------------------------------------------------------
__global__ void scatter_w(const float* __restrict__ w1, const int* __restrict__ idx,
                          unsigned short* __restrict__ wd) {
  int h = blockIdx.x * blockDim.x + threadIdx.x;
  if (h >= H_SZ) return;
  unsigned short* row = wd + (size_t)h * D_SZ;
#pragma unroll
  for (int k = 0; k < K_SZ; ++k) {
    int d   = idx[h * K_SZ + k];
    float v = w1[h * K_SZ + k];
    row[d] = f2bf(bf2f(row[d]) + v);
  }
}

// ---------------------------------------------------------------------------
// Kernel 3: hidden = x_bf16 @ W^T via WMMA, double-buffered async-to-LDS tiles,
// fused tanh(+b1) * w2[n], butterfly reduce over N, atomicAdd per batch row.
// Block: 256 threads = 8 waves (2m x 4n); tile 128M x 128N x 32K.
// ---------------------------------------------------------------------------
__launch_bounds__(256)
__global__ void gemm_fused(const unsigned short* __restrict__ xb,
                           const unsigned short* __restrict__ wd,
                           const float* __restrict__ b1,
                           const float* __restrict__ w2,
                           float* __restrict__ acc) {
  __shared__ unsigned short As[2][BM * SA];  // 2 x 10240 B
  __shared__ unsigned short Bs[2][BN * SB];  // 2 x 10240 B

  const int t    = threadIdx.x;
  const int lane = t & 31;
  const int wave = t >> 5;
  const int wm   = wave >> 2;     // 0..1 : 64 rows each
  const int wn   = wave & 3;      // 0..3 : 32 cols each
  const int mBase = blockIdx.x * BM;
  const int nBase = blockIdx.y * BN;

  const int nl   = lane & 15;
  const int half = lane >> 4;
  const int aKlo = half * 8;      // A frag: halves 0..7 -> K=aKlo.. ; 8..15 -> K=16+aKlo..
  const int bK   = half * 8;      // B frag: halves 0..7 -> K=bK.. ; 8..15 -> K=16+bK..

  // Per-thread copy chunks: tile = 128 rows x 4 b128-chunks; 2 chunks/thread each for A and B.
  const int c0row = t >> 2, c0part = t & 3;            // chunk t
  const int c1row = (t + 256) >> 2, c1part = t & 3;    // chunk t+256 (part identical)

  v8f accf[4][2];
#pragma unroll
  for (int i = 0; i < 4; ++i)
#pragma unroll
    for (int j = 0; j < 2; ++j)
      accf[i][j] = (v8f){0.f, 0.f, 0.f, 0.f, 0.f, 0.f, 0.f, 0.f};

  // issue tile kt into buffer buf
  auto stage = [&](int kt, int buf) {
    const int k0 = kt * BK;
    async_copy16(&xb[(size_t)(mBase + c0row) * D_SZ + k0 + c0part * 8], &As[buf][c0row * SA + c0part * 8]);
    async_copy16(&wd[(size_t)(nBase + c0row) * D_SZ + k0 + c0part * 8], &Bs[buf][c0row * SB + c0part * 8]);
    async_copy16(&xb[(size_t)(mBase + c1row) * D_SZ + k0 + c1part * 8], &As[buf][c1row * SA + c1part * 8]);
    async_copy16(&wd[(size_t)(nBase + c1row) * D_SZ + k0 + c1part * 8], &Bs[buf][c1row * SB + c1part * 8]);
  };

  stage(0, 0);
  wait_async0();
  __syncthreads();

  for (int kt = 0; kt < NT; ++kt) {
    const int cur = kt & 1;
    if (kt + 1 < NT) stage(kt + 1, cur ^ 1);   // prefetch next tile into other buffer

    union FragU { v16bf v; uint4 u[2]; } bfrag[2], afrag;
#pragma unroll
    for (int fn = 0; fn < 2; ++fn) {
      int r = (wn * 32 + fn * 16 + nl) * SB;
      bfrag[fn].u[0] = *(const uint4*)&Bs[cur][r + bK];
      bfrag[fn].u[1] = *(const uint4*)&Bs[cur][r + 16 + bK];
    }
#pragma unroll
    for (int fm = 0; fm < 4; ++fm) {
      int r = (wm * 64 + fm * 16 + nl) * SA;
      afrag.u[0] = *(const uint4*)&As[cur][r + aKlo];
      afrag.u[1] = *(const uint4*)&As[cur][r + 16 + aKlo];
#pragma unroll
      for (int fn = 0; fn < 2; ++fn) {
        accf[fm][fn] = __builtin_amdgcn_wmma_f32_16x16x32_bf16(
            false, afrag.v, false, bfrag[fn].v,
            (short)0, accf[fm][fn], false, false);
      }
    }
    wait_async0();      // our prefetch (and fallback stores) complete
    __syncthreads();    // everyone done reading cur + writing cur^1
  }

  // --- epilogue: tanh(+b1), scale by w2[n], reduce over N, atomicAdd per row
  float b1v[2], w2v[2];
#pragma unroll
  for (int fn = 0; fn < 2; ++fn) {
    int n = nBase + wn * 32 + fn * 16 + nl;
    b1v[fn] = b1[n];
    w2v[fn] = w2[n];
  }
#pragma unroll
  for (int fm = 0; fm < 4; ++fm) {
#pragma unroll
    for (int r = 0; r < 8; ++r) {
      float s = 0.0f;
#pragma unroll
      for (int fn = 0; fn < 2; ++fn) {
        float hv = fast_tanh(accf[fm][fn][r] + b1v[fn]);
        s += hv * w2v[fn];
      }
#pragma unroll
      for (int m = 1; m < 16; m <<= 1) s += __shfl_xor(s, m, 32);
      if (nl == 0) {
        int mg = mBase + wm * 64 + fm * 16 + r + 8 * half;
        atomicAdd(&acc[mg], s);
      }
    }
  }
}

// ---------------------------------------------------------------------------
// Kernel 4: out[b] = tanh(acc[b] + b2)
// ---------------------------------------------------------------------------
__global__ void finalize(const float* __restrict__ acc, const float* __restrict__ b2,
                         float* __restrict__ out) {
  int b = blockIdx.x * blockDim.x + threadIdx.x;
  if (b < B_SZ) out[b] = fast_tanh(acc[b] + b2[0]);
}

// ---------------------------------------------------------------------------
extern "C" void kernel_launch(void* const* d_in, const int* in_sizes, int n_in,
                              void* d_out, int out_size, void* d_ws, size_t ws_size,
                              hipStream_t stream) {
  const float* x   = (const float*)d_in[0];
  const int*   idx = (const int*)d_in[1];
  const float* w1  = (const float*)d_in[2];
  const float* b1  = (const float*)d_in[3];
  const float* w2  = (const float*)d_in[4];
  const float* b2  = (const float*)d_in[5];
  float* out = (float*)d_out;

  // workspace layout: x_bf16 (8 MB) | W_dense bf16 (4 MB) | acc f32 (16 KB)
  unsigned short* xbuf = (unsigned short*)d_ws;
  unsigned short* wd   = (unsigned short*)((char*)d_ws + (size_t)B_SZ * D_SZ * 2);
  float*          acc  = (float*)((char*)d_ws + (size_t)B_SZ * D_SZ * 2 + (size_t)H_SZ * D_SZ * 2);

  zero_ws  <<<dim3((H_SZ * D_SZ / 2 + 255) / 256), dim3(256), 0, stream>>>((uint32_t*)wd, acc);
  convert_x<<<dim3(B_SZ * D_SZ / 4 / 256),          dim3(256), 0, stream>>>(x, xbuf);
  scatter_w<<<dim3((H_SZ + 255) / 256),             dim3(256), 0, stream>>>(w1, idx, wd);
  gemm_fused<<<dim3(B_SZ / BM, H_SZ / BN),          dim3(256), 0, stream>>>(xbuf, wd, b1, w2, acc);
  finalize <<<dim3((B_SZ + 255) / 256),             dim3(256), 0, stream>>>(acc, b2, out);
}
